// FashionMNISTManualBNDropout_53815940219270
// MI455X (gfx1250) — matrix operand order repaired and checked by reading, not compiled, not run on hardware
//
#include <hip/hip_runtime.h>
#include <hip/hip_bf16.h>

// MLP: GEMM1 -> BN/ReLU/dropout -> GEMM2 -> BN/ReLU/dropout -> GEMM3
// GEMMs use v_wmma_f32_16x16x32_bf16 with 2-term bf16 split of fp32 operands
// (Ahi*Bhi + Alo*Bhi + Ahi*Blo) for near-fp32 accuracy at bf16 matrix rates.

typedef __attribute__((ext_vector_type(16))) __bf16    v16bf;
typedef __attribute__((ext_vector_type(8)))  float     v8f;
typedef __attribute__((ext_vector_type(4)))  float     v4f;
typedef __attribute__((ext_vector_type(8)))  unsigned  u8v;

#define DROP_SCALE (1.0f / 0.7f)

// Fragment stored as 8 packed dwords (16 bf16) for hi and lo split terms.
struct Frag { u8v hi; u8v lo; };

// Load 8 contiguous floats (two aligned 16B vectors).
__device__ __forceinline__ void load8(const float* __restrict__ p, float o[8]) {
    v4f a = *(const v4f*)p;
    v4f b = *(const v4f*)(p + 4);
    o[0]=a.x; o[1]=a.y; o[2]=a.z; o[3]=a.w;
    o[4]=b.x; o[5]=b.y; o[6]=b.z; o[7]=b.w;
}

__device__ __forceinline__ void zero8(float o[8]) {
#pragma unroll
    for (int i = 0; i < 8; ++i) o[i] = 0.0f;
}

// Split 8 fp32 into packed bf16 hi (truncation, exact residual) + bf16 lo,
// writing packed words [w0, w0+4) of the fragment halves.
// v_perm_b32 packs the top halves of two dwords: result = {u1.b3,u1.b2,u0.b3,u0.b2}
__device__ __forceinline__ void split8(const float v[8], u8v& hi, u8v& lo, int w0) {
#pragma unroll
    for (int p = 0; p < 4; ++p) {
        unsigned u0 = __builtin_bit_cast(unsigned, v[2 * p]);
        unsigned u1 = __builtin_bit_cast(unsigned, v[2 * p + 1]);
        hi[w0 + p] = __builtin_amdgcn_perm(u1, u0, 0x07060302u);
        float r0 = v[2 * p]     - __builtin_bit_cast(float, u0 & 0xFFFF0000u);
        float r1 = v[2 * p + 1] - __builtin_bit_cast(float, u1 & 0xFFFF0000u);
        lo[w0 + p] = __builtin_amdgcn_perm(__builtin_bit_cast(unsigned, r1),
                                           __builtin_bit_cast(unsigned, r0), 0x07060302u);
    }
}

// BN affine + ReLU + dropout-mask on 8 consecutive-k elements of a row.
__device__ __forceinline__ void transform8(float o[8],
                                           const float* __restrict__ sc,
                                           const float* __restrict__ sh,
                                           const float* __restrict__ mrow, int kb) {
#pragma unroll
    for (int i = 0; i < 8; ++i) {
        float t = fmaf(o[i], sc[kb + i], sh[kb + i]);
        t = fmaxf(t, 0.0f);
        o[i] = t * mrow[kb + i] * DROP_SCALE;
    }
}

__device__ __forceinline__ v8f wmma_split3(const Frag& a, const Frag& b, v8f c) {
    v16bf ah = __builtin_bit_cast(v16bf, a.hi);
    v16bf al = __builtin_bit_cast(v16bf, a.lo);
    v16bf bh = __builtin_bit_cast(v16bf, b.hi);
    v16bf bl = __builtin_bit_cast(v16bf, b.lo);
    c = __builtin_amdgcn_wmma_f32_16x16x32_bf16(false, ah, false, bh, (short)0, c, false, false);
    c = __builtin_amdgcn_wmma_f32_16x16x32_bf16(false, al, false, bh, (short)0, c, false, false);
    c = __builtin_amdgcn_wmma_f32_16x16x32_bf16(false, ah, false, bl, (short)0, c, false, false);
    return c;
}

// C[B,N] = op(A)[B,K] @ W[N,K]^T + bias.  Block tile 128x64, 8 waves, each wave
// does a 2x2 grid of 16x16 WMMA tiles.  TRANS=true applies BN/ReLU/mask to A.
// Main loop over K&~31 is unpredicated; a single 16-wide tail (K%32==16,
// K%16==0, only layer 1) handles the remainder.
template <bool TRANS>
__launch_bounds__(256)
__global__ void gemm_bn_kernel(const float* __restrict__ A, const float* __restrict__ W,
                               const float* __restrict__ bias,
                               const float* __restrict__ bnscale, const float* __restrict__ bnshift,
                               const float* __restrict__ mask,
                               float* __restrict__ C, int K, int N) {
    const int tid  = threadIdx.x;
    const int lane = tid & 31;
    const int wave = tid >> 5;
    const int m16  = lane & 15;
    const int h    = lane >> 4;      // half-wave select
    const int wm   = wave & 3;       // 0..3 -> row subtile
    const int wn   = wave >> 2;      // 0..1 -> col subtile
    const size_t row0 = (size_t)blockIdx.y * 128 + wm * 32;
    const int    col0 = blockIdx.x * 64 + wn * 32;
    const int    Kmain = K & ~31;

    v8f acc[2][2] = {};

    for (int k0 = 0; k0 < Kmain; k0 += 32) {
        Frag afr[2], bfr[2];

        // B fragments: lane (n=m16, h) holds W[n][k0+h*8..+7] and W[n][k0+16+h*8..+7]
#pragma unroll
        for (int ni = 0; ni < 2; ++ni) {
            const float* wp = W + (size_t)(col0 + ni * 16 + m16) * K + k0 + h * 8;
            float c0[8], c1[8];
            load8(wp,     c0);
            load8(wp + 16, c1);
            split8(c0, bfr[ni].hi, bfr[ni].lo, 0);
            split8(c1, bfr[ni].hi, bfr[ni].lo, 4);
        }

        // A fragments: lane (m=m16, h) holds A[m][k0+h*8..+7] and A[m][k0+16+h*8..+7]
#pragma unroll
        for (int mi = 0; mi < 2; ++mi) {
            const size_t row = row0 + mi * 16 + m16;
            const float* ap = A + row * K;
            const int ka = k0 + h * 8;
            float c0[8], c1[8];
            load8(ap + ka,      c0);
            load8(ap + ka + 16, c1);
            if (k0 + 32 < K) __builtin_prefetch(ap + ka + 32, 0, 3);
            if (TRANS) {
                const float* mrow = mask + row * K;
                transform8(c0, bnscale, bnshift, mrow, ka);
                transform8(c1, bnscale, bnshift, mrow, ka + 16);
            }
            split8(c0, afr[mi].hi, afr[mi].lo, 0);
            split8(c1, afr[mi].hi, afr[mi].lo, 4);
        }

#pragma unroll
        for (int mi = 0; mi < 2; ++mi)
#pragma unroll
            for (int ni = 0; ni < 2; ++ni)
                acc[mi][ni] = wmma_split3(afr[mi], bfr[ni], acc[mi][ni]);
    }

    // 16-wide K tail (layer 1 only: K=784).  A-chunk0 valid, A-chunk1 zero.
    // B fragment: chunk0 = W[n][Kmain+h*8..+7], chunk1 zero.
    if (!TRANS && Kmain < K) {
        Frag afr[2], bfr[2];
#pragma unroll
        for (int ni = 0; ni < 2; ++ni) {
            float c0[8], c1[8];
            {
                const float* wp = W + (size_t)(col0 + ni * 16 + m16) * K + Kmain + h * 8;
                load8(wp,     c0);
                zero8(c1);
            }
            split8(c0, bfr[ni].hi, bfr[ni].lo, 0);
            split8(c1, bfr[ni].hi, bfr[ni].lo, 4);
        }
#pragma unroll
        for (int mi = 0; mi < 2; ++mi) {
            const size_t row = row0 + mi * 16 + m16;
            float c0[8], c1[8];
            load8(A + row * K + Kmain + h * 8, c0);
            zero8(c1);
            split8(c0, afr[mi].hi, afr[mi].lo, 0);
            split8(c1, afr[mi].hi, afr[mi].lo, 4);
        }
#pragma unroll
        for (int mi = 0; mi < 2; ++mi)
#pragma unroll
            for (int ni = 0; ni < 2; ++ni)
                acc[mi][ni] = wmma_split3(afr[mi], bfr[ni], acc[mi][ni]);
    }

    // C/D layout: VGPR j -> row = row0 + mi*16 + h*8 + j, col = col0 + ni*16 + m16
#pragma unroll
    for (int mi = 0; mi < 2; ++mi)
#pragma unroll
        for (int ni = 0; ni < 2; ++ni) {
            const int col = col0 + ni * 16 + m16;
            const float bv = bias[col];
#pragma unroll
            for (int j = 0; j < 8; ++j) {
                const size_t row = row0 + mi * 16 + h * 8 + j;
                C[row * N + col] = acc[mi][ni][j] + bv;
            }
        }
}

// Final layer: C[B,10] = T(A) @ W3[10,512]^T + b3.  W3 is staged zero-padded to
// 16x512 in LDS once per block (32KB), so the k-loop B reads are uniform
// ds_load_b128 with no divergence.  Block = 8 waves x 32 rows = 256 rows.
__launch_bounds__(256)
__global__ void gemm_out_kernel(const float* __restrict__ A, const float* __restrict__ W,
                                const float* __restrict__ bias,
                                const float* __restrict__ bnscale, const float* __restrict__ bnshift,
                                const float* __restrict__ mask,
                                float* __restrict__ C, int K, int Nvalid) {
    __shared__ float wlds[16 * 512];

    const int tid  = threadIdx.x;
    const int lane = tid & 31;
    const int wave = tid >> 5;
    const int m16  = lane & 15;
    const int h    = lane >> 4;
    const size_t row0 = (size_t)blockIdx.x * 256 + wave * 32;

    // Stage zero-padded W3 into LDS.
    for (int i = tid; i < 16 * 512; i += 256) {
        const int n = i >> 9;
        wlds[i] = (n < Nvalid) ? W[(size_t)n * K + (i & 511)] : 0.0f;
    }
    __syncthreads();

    v8f acc[2] = {};

    for (int k0 = 0; k0 < K; k0 += 32) {
        Frag bfr, afr[2];
        {
            const float* wp = wlds + m16 * 512 + k0 + h * 8;
            float c0[8], c1[8];
            load8(wp,     c0);
            load8(wp + 16, c1);
            split8(c0, bfr.hi, bfr.lo, 0);
            split8(c1, bfr.hi, bfr.lo, 4);
        }
#pragma unroll
        for (int mi = 0; mi < 2; ++mi) {
            const size_t row = row0 + mi * 16 + m16;
            const float* ap   = A + row * K;
            const float* mrow = mask + row * K;
            const int ka = k0 + h * 8;
            float c0[8], c1[8];
            load8(ap + ka,      c0);
            load8(ap + ka + 16, c1);
            if (k0 + 32 < K) __builtin_prefetch(ap + ka + 32, 0, 3);
            transform8(c0, bnscale, bnshift, mrow, ka);
            transform8(c1, bnscale, bnshift, mrow, ka + 16);
            split8(c0, afr[mi].hi, afr[mi].lo, 0);
            split8(c1, afr[mi].hi, afr[mi].lo, 4);
        }
#pragma unroll
        for (int mi = 0; mi < 2; ++mi)
            acc[mi] = wmma_split3(afr[mi], bfr, acc[mi]);
    }

    const int col = m16;
    if (col < Nvalid) {
        const float bv = bias[col];
#pragma unroll
        for (int mi = 0; mi < 2; ++mi)
#pragma unroll
            for (int j = 0; j < 8; ++j) {
                const size_t row = row0 + mi * 16 + h * 8 + j;
                C[row * Nvalid + col] = acc[mi][j] + bv;
            }
    }
}

// Per-column partial sums over a 256-row slab; coalesced reads, atomic merge.
__global__ void stats_partial(const float* __restrict__ H, float* __restrict__ sum,
                              float* __restrict__ sq) {
    const int c0 = threadIdx.x;                  // 0..255
    const size_t r0 = (size_t)blockIdx.x * 256;
    float s0 = 0.f, s1 = 0.f, q0 = 0.f, q1 = 0.f;
    for (int r = 0; r < 256; ++r) {
        const float* row = H + (r0 + r) * 512;
        const float a = row[c0];
        const float b = row[c0 + 256];
        s0 += a; q0 += a * a;
        s1 += b; q1 += b * b;
    }
    atomicAdd(&sum[c0], s0);       atomicAdd(&sq[c0], q0);
    atomicAdd(&sum[c0 + 256], s1); atomicAdd(&sq[c0 + 256], q1);
}

__global__ void stats_final(const float* __restrict__ sum, const float* __restrict__ sq,
                            const float* __restrict__ gamma, const float* __restrict__ beta,
                            float* __restrict__ scale, float* __restrict__ shift) {
    const int c = blockIdx.x * blockDim.x + threadIdx.x;
    if (c >= 512) return;
    const float inv_b = 1.0f / 32768.0f;
    const float mean = sum[c] * inv_b;
    const float var  = sq[c] * inv_b - mean * mean;
    const float sc   = gamma[c] * rsqrtf(var + 1e-5f);
    scale[c] = sc;
    shift[c] = beta[c] - mean * sc;
}

__global__ void zero_kernel(float* __restrict__ p, int n) {
    const int i = blockIdx.x * blockDim.x + threadIdx.x;
    if (i < n) p[i] = 0.0f;
}

extern "C" void kernel_launch(void* const* d_in, const int* in_sizes, int n_in,
                              void* d_out, int out_size, void* d_ws, size_t ws_size,
                              hipStream_t stream) {
    const float* x      = (const float*)d_in[0];
    const float* W1     = (const float*)d_in[1];
    const float* b1     = (const float*)d_in[2];
    const float* W2     = (const float*)d_in[3];
    const float* b2     = (const float*)d_in[4];
    const float* W3     = (const float*)d_in[5];
    const float* b3     = (const float*)d_in[6];
    const float* gamma1 = (const float*)d_in[7];
    const float* beta1  = (const float*)d_in[8];
    const float* gamma2 = (const float*)d_in[9];
    const float* beta2  = (const float*)d_in[10];
    const float* mask1  = (const float*)d_in[11];
    const float* mask2  = (const float*)d_in[12];
    float* out = (float*)d_out;

    constexpr int B = 32768, DIN = 784, H = 512, DOUT = 10;

    float* h1 = (float*)d_ws;                     // B*H fp32
    float* h2 = h1 + (size_t)B * H;               // B*H fp32
    float* st = h2 + (size_t)B * H;               // stats scratch
    float* sum1 = st,        * sq1   = st + 512;
    float* sum2 = st + 1024, * sq2   = st + 1536;
    float* scale1 = st + 2048, * shift1 = st + 2560;
    float* scale2 = st + 3072, * shift2 = st + 3584;

    zero_kernel<<<2, 1024, 0, stream>>>(st, 2048);

    // Layer 1: h1 = x @ W1^T + b1
    gemm_bn_kernel<false><<<dim3(H / 64, B / 128), 256, 0, stream>>>(
        x, W1, b1, nullptr, nullptr, nullptr, h1, DIN, H);

    stats_partial<<<B / 256, 256, 0, stream>>>(h1, sum1, sq1);
    stats_final<<<2, 256, 0, stream>>>(sum1, sq1, gamma1, beta1, scale1, shift1);

    // Layer 2: h2 = (relu(bn(h1)) * mask1 / 0.7) @ W2^T + b2  (BN fused into A load)
    gemm_bn_kernel<true><<<dim3(H / 64, B / 128), 256, 0, stream>>>(
        h1, W2, b2, scale1, shift1, mask1, h2, H, H);

    stats_partial<<<B / 256, 256, 0, stream>>>(h2, sum2, sq2);
    stats_final<<<2, 256, 0, stream>>>(sum2, sq2, gamma2, beta2, scale2, shift2);

    // Layer 3: out = (relu(bn(h2)) * mask2 / 0.7) @ W3^T + b3
    gemm_out_kernel<<<B / 256, 256, 0, stream>>>(
        h2, W3, b3, scale2, shift2, mask2, out, H, DOUT);
}
